// MambaNet_17377437680258
// MI455X (gfx1250) — matrix operand order repaired and checked by reading, not compiled, not run on hardware
//
#include <hip/hip_runtime.h>
#include <hip/hip_bf16.h>
#include <math.h>

// Problem constants (from reference)
#define B_SZ 4
#define L_SZ 2048
#define D_SZ 1024
#define E_SZ 2048          // d_inner
#define N_SZ 8             // d_state
#define K_CONV 4
#define R_SZ 64            // dt_rank
#define BL (B_SZ * L_SZ)   // 8192 rows
#define DBC (R_SZ + 2 * N_SZ)  // 80

typedef __attribute__((ext_vector_type(16))) __bf16 v16bf;
typedef __attribute__((ext_vector_type(8)))  float  v8f;

// ---------------------------------------------------------------------------
// Elementwise helpers
// ---------------------------------------------------------------------------
__global__ void k_cvt_bf16(const float* __restrict__ in, __bf16* __restrict__ out, int n) {
    int i = blockIdx.x * blockDim.x + threadIdx.x;
    if (i < n) out[i] = (__bf16)in[i];
}

__global__ void k_negexp(const float* __restrict__ in, float* __restrict__ out, int n) {
    int i = blockIdx.x * blockDim.x + threadIdx.x;
    if (i < n) out[i] = -__expf(in[i]);
}

// Depthwise causal conv1d (K=4) + bias + SiLU over the x half of xz.
__global__ void k_conv_silu(const float* __restrict__ xz,
                            const float* __restrict__ cw,   // (E, 1, K)
                            const float* __restrict__ cb,   // (E)
                            float* __restrict__ xf,
                            __bf16* __restrict__ xb) {
    int idx = blockIdx.x * blockDim.x + threadIdx.x;        // over B*L*E
    if (idx >= BL * E_SZ) return;
    int e  = idx % E_SZ;
    int bl = idx / E_SZ;
    int l  = bl % L_SZ;
    int b  = bl / L_SZ;
    float acc = cb[e];
#pragma unroll
    for (int j = 0; j < K_CONV; j++) {
        int lj = l + j - (K_CONV - 1);
        if (lj >= 0)
            acc += cw[e * K_CONV + j] *
                   xz[((size_t)(b * L_SZ + lj)) * (2 * E_SZ) + e];
    }
    float s = acc / (1.f + __expf(-acc));                   // SiLU
    xf[idx] = s;
    xb[idx] = (__bf16)s;
}

// ---------------------------------------------------------------------------
// Tiled bf16 WMMA GEMM: C[M,N] = A[M,K] @ W[N,K]^T   (f32 accumulate)
// Block tile 128x128, BK=32, LDS double-buffered (ping-pong, 32 KB).
// 8 waves: 4 (M) x 2 (N); each wave 32x64 (2x4 WMMA -> 8 v_wmma / k-step).
// GUARD: predicate N-edge (only x_proj, N=80). DUALOUT: fused bf16 mirror.
// ---------------------------------------------------------------------------
#define BM 128
#define BN 128
#define BK 32

template <bool GUARD, bool DUALOUT>
__global__ __launch_bounds__(256) void k_gemm_bf16(
    const __bf16* __restrict__ A, int lda,
    const __bf16* __restrict__ W, int ldw,
    float* __restrict__ C, int ldc,
    __bf16* __restrict__ Cb,         // bf16 copy of C (used iff DUALOUT)
    int M, int N, int K)
{
    __shared__ __bf16 sA[2][BM * BK];   // 2 x 8 KB
    __shared__ __bf16 sW[2][BN * BK];   // 2 x 8 KB

    const int tid  = threadIdx.x;
    const int lane = tid & 31;
    const int wave = tid >> 5;
    const int wm   = wave & 3;       // 0..3  (M direction, 32 rows each)
    const int wn   = wave >> 2;      // 0..1  (N direction, 64 cols each)
    const int half = (lane >> 4) & 1;
    const int l16  = lane & 15;
    const int blockRow = blockIdx.y * BM;
    const int blockCol = blockIdx.x * BN;

    // Static staging map: 128-bit vectors (8 bf16) per load.
    // A tile: 128 rows x 4 vecs = 512 vecs -> 2 per thread; same shape for W.
    const int ar0 = tid >> 2,          ac0 = tid & 3;
    const int ar1 = (tid + 256) >> 2,  ac1 = tid & 3;

    const __bf16* gA0 = A + (size_t)(blockRow + ar0) * lda + ac0 * 8;
    const __bf16* gA1 = A + (size_t)(blockRow + ar1) * lda + ac1 * 8;
    const bool w0ok = !GUARD || (blockCol + ar0 < N);
    const bool w1ok = !GUARD || (blockCol + ar1 < N);
    const __bf16* gW0 = W + (size_t)(blockCol + (w0ok ? ar0 : 0)) * ldw + ac0 * 8;
    const __bf16* gW1 = W + (size_t)(blockCol + (w1ok ? ar1 : 0)) * ldw + ac1 * 8;

    v8f acc[2][4];
#pragma unroll
    for (int i = 0; i < 2; i++)
#pragma unroll
        for (int j = 0; j < 4; j++)
#pragma unroll
            for (int r = 0; r < 8; r++) acc[i][j][r] = 0.f;

    const int nk = K / BK;

    // ---- prologue: stage tile 0 into buffer 0
    {
        uint4 a0 = *(const uint4*)(gA0);
        uint4 a1 = *(const uint4*)(gA1);
        uint4 w0 = make_uint4(0u, 0u, 0u, 0u), w1 = make_uint4(0u, 0u, 0u, 0u);
        if (w0ok) w0 = *(const uint4*)(gW0);
        if (w1ok) w1 = *(const uint4*)(gW1);
        *(uint4*)&sA[0][ar0 * BK + ac0 * 8] = a0;
        *(uint4*)&sA[0][ar1 * BK + ac1 * 8] = a1;
        *(uint4*)&sW[0][ar0 * BK + ac0 * 8] = w0;
        *(uint4*)&sW[0][ar1 * BK + ac1 * 8] = w1;
    }
    __syncthreads();

    for (int kt = 0; kt < nk; kt++) {
        const int cur = kt & 1;
        const bool have_next = (kt + 1 < nk);

        // Issue next tile's global loads; they overlap this tile's WMMA math.
        uint4 na0, na1, nw0, nw1;
        if (have_next) {
            const int ko = (kt + 1) * BK;
            na0 = *(const uint4*)(gA0 + ko);
            na1 = *(const uint4*)(gA1 + ko);
            nw0 = make_uint4(0u, 0u, 0u, 0u);
            nw1 = make_uint4(0u, 0u, 0u, 0u);
            if (w0ok) nw0 = *(const uint4*)(gW0 + ko);
            if (w1ok) nw1 = *(const uint4*)(gW1 + ko);
            if (kt + 2 < nk) {   // prefetch k+2 (global_prefetch_b8)
                __builtin_prefetch(gA0 + (kt + 2) * BK, 0, 3);
                __builtin_prefetch(gW0 + (kt + 2) * BK, 0, 3);
            }
        }

        // Fragments per CDNA5 ISA layouts.
        // A 16x32 bf16: lane<16 row M=l16 holds K0..7 (VGPR0-3, pair-packed)
        // and K16..23 (VGPR4-7); lane>=16 holds K8..15 / K24..31.
        v16bf afr[2], bfr[4];
#pragma unroll
        for (int sm = 0; sm < 2; sm++) {
            int rbase = (wm * 32 + sm * 16 + l16) * BK;
#pragma unroll
            for (int i = 0; i < 16; i++) {
                int j = i >> 1, p = i & 1;
                int kk = (j < 4) ? (half * 8 + j * 2 + p)
                                 : (16 + half * 8 + (j - 4) * 2 + p);
                afr[sm][i] = sA[cur][rbase + kk];
            }
        }
        // B 32x16 bf16: same K order as A: lane<16 holds K0..7 / K16..23; lane>=16 holds K8..15 / K24..31.
#pragma unroll
        for (int sn = 0; sn < 4; sn++) {
            int nbase = (wn * 64 + sn * 16 + l16) * BK;
#pragma unroll
            for (int i = 0; i < 16; i++) {
                int v = i >> 1, p = i & 1;
                int kk = (v < 4) ? (half * 8 + v * 2 + p)
                                 : (16 + half * 8 + (v - 4) * 2 + p);
                bfr[sn][i] = sW[cur][nbase + kk];
            }
        }
#pragma unroll
        for (int sm = 0; sm < 2; sm++)
#pragma unroll
            for (int sn = 0; sn < 4; sn++)
                acc[sm][sn] = __builtin_amdgcn_wmma_f32_16x16x32_bf16(
                    false, afr[sm], false, bfr[sn],
                    (short)0, acc[sm][sn], false, false);

        // Stage next tile into the other buffer (no race: readers use `cur`).
        if (have_next) {
            const int nxt = 1 - cur;
            *(uint4*)&sA[nxt][ar0 * BK + ac0 * 8] = na0;
            *(uint4*)&sA[nxt][ar1 * BK + ac1 * 8] = na1;
            *(uint4*)&sW[nxt][ar0 * BK + ac0 * 8] = nw0;
            *(uint4*)&sW[nxt][ar1 * BK + ac1 * 8] = nw1;
        }
        __syncthreads();
    }

    // Store per C/D layout: VGPR r, lane<16 -> M=r, N=lane; lane>=16 -> M=8+r.
#pragma unroll
    for (int sm = 0; sm < 2; sm++)
#pragma unroll
        for (int sn = 0; sn < 4; sn++) {
            int gr0 = blockRow + wm * 32 + sm * 16 + half * 8;
            int gc  = blockCol + wn * 64 + sn * 16 + l16;
            if (!GUARD || gc < N) {
#pragma unroll
                for (int r = 0; r < 8; r++) {
                    float v = acc[sm][sn][r];
                    C[(size_t)(gr0 + r) * ldc + gc] = v;
                    if (DUALOUT) Cb[(size_t)(gr0 + r) * ldc + gc] = (__bf16)v;
                }
            }
        }
}

// ---------------------------------------------------------------------------
// Selective scan: one thread per (b, e) channel, N=8 states in registers.
// Fuses softplus(dt+bias), dA=exp(delta*A), skip u*D, and silu(z) gate.
// B/C staged through LDS in chunks of 64 timesteps.
// ---------------------------------------------------------------------------
__global__ __launch_bounds__(256) void k_scan(
    const float* __restrict__ dtout,  // (BL, E)  dt @ Wdt^T (pre-bias)
    const float* __restrict__ dtb,    // (E)
    const float* __restrict__ dbc,    // (BL, 80) [dt(64) | B(8) | C(8)]
    const float* __restrict__ xf,     // (BL, E)  conv+silu output u
    const float* __restrict__ xz,     // (BL, 2E) z = cols [E, 2E)
    const float* __restrict__ Aneg,   // (E, N)   A = -exp(A_log)
    const float* __restrict__ Dp,     // (E)
    __bf16* __restrict__ yb)          // (BL, E)  gated output, bf16
{
    __shared__ float sBC[64 * 16];
    const int b = blockIdx.y;
    const int e = blockIdx.x * 256 + threadIdx.x;

    float h[N_SZ], Av[N_SZ];
#pragma unroll
    for (int n = 0; n < N_SZ; n++) { h[n] = 0.f; Av[n] = Aneg[e * N_SZ + n]; }
    const float bias  = dtb[e];
    const float dskip = Dp[e];

    for (int c = 0; c < L_SZ / 64; c++) {
        const int t0 = c * 64;
        __syncthreads();
        for (int p = threadIdx.x; p < 64 * 16; p += 256) {
            int tl = p >> 4, cc = p & 15;
            sBC[p] = dbc[((size_t)(b * L_SZ + t0 + tl)) * DBC + R_SZ + cc];
        }
        __syncthreads();
        for (int tl = 0; tl < 64; tl++) {
            const size_t row = (size_t)(b * L_SZ + t0 + tl);
            float draw  = dtout[row * E_SZ + e] + bias;
            float delta = (draw > 20.f) ? draw : log1pf(__expf(draw));
            float u  = xf[row * E_SZ + e];
            float du = delta * u;
            const float* Bt = &sBC[tl * 16];
            float y = 0.f;
#pragma unroll
            for (int n = 0; n < N_SZ; n++) {
                h[n] = h[n] * __expf(delta * Av[n]) + du * Bt[n];
                y += h[n] * Bt[8 + n];
            }
            y += u * dskip;
            float z = xz[row * (2 * E_SZ) + E_SZ + e];
            y *= z / (1.f + __expf(-z));                    // * silu(z)
            yb[row * E_SZ + e] = (__bf16)y;
        }
    }
}

// ---------------------------------------------------------------------------
// RMSNorm over D=1024 per row -> final f32 output.
// ---------------------------------------------------------------------------
__global__ __launch_bounds__(256) void k_rmsnorm(const float* __restrict__ in,
                                                 const float* __restrict__ w,
                                                 float* __restrict__ out) {
    __shared__ float red[256];
    const int row = blockIdx.x;
    const float* r = in + (size_t)row * D_SZ;
    float s = 0.f;
    for (int i = threadIdx.x; i < D_SZ; i += 256) { float v = r[i]; s += v * v; }
    red[threadIdx.x] = s;
    __syncthreads();
    for (int st = 128; st > 0; st >>= 1) {
        if (threadIdx.x < st) red[threadIdx.x] += red[threadIdx.x + st];
        __syncthreads();
    }
    float scale = rsqrtf(red[0] / (float)D_SZ + 1e-5f);
    for (int i = threadIdx.x; i < D_SZ; i += 256)
        out[(size_t)row * D_SZ + i] = r[i] * scale * w[i];
}

// ---------------------------------------------------------------------------
// Host orchestration
// ---------------------------------------------------------------------------
extern "C" void kernel_launch(void* const* d_in, const int* in_sizes, int n_in,
                              void* d_out, int out_size, void* d_ws, size_t ws_size,
                              hipStream_t stream) {
    (void)in_sizes; (void)n_in; (void)out_size; (void)ws_size;

    const float* inp    = (const float*)d_in[0];   // (B,L,D)
    const float* w_in   = (const float*)d_in[1];   // (2E, D)
    const float* conv_w = (const float*)d_in[2];   // (E,1,K)
    const float* conv_b = (const float*)d_in[3];   // (E)
    const float* w_xp   = (const float*)d_in[4];   // (80, E)
    const float* w_dt   = (const float*)d_in[5];   // (E, R)
    const float* dt_b   = (const float*)d_in[6];   // (E)
    const float* A_log  = (const float*)d_in[7];   // (E, N)
    const float* D_par  = (const float*)d_in[8];   // (E)
    const float* w_out  = (const float*)d_in[9];   // (D, E)
    const float* rms_w  = (const float*)d_in[10];  // (D)
    float* out = (float*)d_out;

    // Workspace carving (256B aligned)
    char*  base = (char*)d_ws;
    size_t off  = 0;
    auto alloc = [&](size_t bytes) -> char* {
        off = (off + 255) & ~(size_t)255;
        char* p = base + off;
        off += bytes;
        return p;
    };
    __bf16* inp_b   = (__bf16*)alloc((size_t)BL * D_SZ * 2);
    __bf16* w_in_b  = (__bf16*)alloc((size_t)2 * E_SZ * D_SZ * 2);
    __bf16* w_xp_b  = (__bf16*)alloc((size_t)DBC * E_SZ * 2);
    __bf16* w_dt_b  = (__bf16*)alloc((size_t)E_SZ * R_SZ * 2);
    __bf16* w_out_b = (__bf16*)alloc((size_t)D_SZ * E_SZ * 2);
    float*  xz      = (float*) alloc((size_t)BL * 2 * E_SZ * 4);
    float*  xf      = (float*) alloc((size_t)BL * E_SZ * 4);
    __bf16* xb      = (__bf16*)alloc((size_t)BL * E_SZ * 2);
    float*  dbc     = (float*) alloc((size_t)BL * DBC * 4);
    __bf16* dbc_b   = (__bf16*)alloc((size_t)BL * DBC * 2);
    float*  dtout   = (float*) alloc((size_t)BL * E_SZ * 4);
    float*  Aneg    = (float*) alloc((size_t)E_SZ * N_SZ * 4);
    __bf16* yb      = (__bf16*)alloc((size_t)BL * E_SZ * 2);
    float*  obuf    = (float*) alloc((size_t)BL * D_SZ * 4);

    const int TPB = 256;
    auto blocks = [](size_t n) { return (unsigned)((n + 255) / 256); };

    // 0) Converts + A precompute
    k_cvt_bf16<<<blocks((size_t)BL * D_SZ), TPB, 0, stream>>>(inp,  inp_b,   BL * D_SZ);
    k_cvt_bf16<<<blocks((size_t)2 * E_SZ * D_SZ), TPB, 0, stream>>>(w_in,  w_in_b,  2 * E_SZ * D_SZ);
    k_cvt_bf16<<<blocks((size_t)DBC * E_SZ), TPB, 0, stream>>>(w_xp,  w_xp_b,  DBC * E_SZ);
    k_cvt_bf16<<<blocks((size_t)E_SZ * R_SZ), TPB, 0, stream>>>(w_dt,  w_dt_b,  E_SZ * R_SZ);
    k_cvt_bf16<<<blocks((size_t)D_SZ * E_SZ), TPB, 0, stream>>>(w_out, w_out_b, D_SZ * E_SZ);
    k_negexp  <<<blocks((size_t)E_SZ * N_SZ), TPB, 0, stream>>>(A_log, Aneg,    E_SZ * N_SZ);

    // 1) in_proj: xz = inp @ w_in^T   (8192 x 4096 x 1024), no edge, f32 only
    k_gemm_bf16<false, false><<<dim3((2 * E_SZ) / BN, BL / BM), TPB, 0, stream>>>(
        inp_b, D_SZ, w_in_b, D_SZ, xz, 2 * E_SZ, nullptr, BL, 2 * E_SZ, D_SZ);

    // 2) depthwise conv + SiLU
    k_conv_silu<<<blocks((size_t)BL * E_SZ), TPB, 0, stream>>>(xz, conv_w, conv_b, xf, xb);

    // 3) x_proj: dbc = x @ w_xp^T     (8192 x 80 x 2048), N-edge + bf16 mirror
    k_gemm_bf16<true, true><<<dim3((DBC + BN - 1) / BN, BL / BM), TPB, 0, stream>>>(
        xb, E_SZ, w_xp_b, E_SZ, dbc, DBC, dbc_b, BL, DBC, E_SZ);

    // 4) dt_proj: dtout = dbc[:, :R] @ w_dt^T  (8192 x 2048 x 64), lda = 80
    k_gemm_bf16<false, false><<<dim3(E_SZ / BN, BL / BM), TPB, 0, stream>>>(
        dbc_b, DBC, w_dt_b, R_SZ, dtout, E_SZ, nullptr, BL, E_SZ, R_SZ);

    // 5) selective scan + skip + gate -> yb (bf16)
    k_scan<<<dim3(E_SZ / 256, B_SZ), TPB, 0, stream>>>(
        dtout, dt_b, dbc, xf, xz, Aneg, D_par, yb);

    // 6) out_proj: obuf = y @ w_out^T (8192 x 1024 x 2048)
    k_gemm_bf16<false, false><<<dim3(D_SZ / BN, BL / BM), TPB, 0, stream>>>(
        yb, E_SZ, w_out_b, E_SZ, obuf, D_SZ, nullptr, BL, D_SZ, E_SZ);

    // 7) RMSNorm -> d_out
    k_rmsnorm<<<dim3(BL), TPB, 0, stream>>>(obuf, rms_w, out);
}
